// ShapePropHead_39539468927321
// MI455X (gfx1250) — matrix operand order repaired and checked by reading, not compiled, not run on hardware
//
#include <hip/hip_runtime.h>
#include <hip/hip_bf16.h>

typedef __attribute__((ext_vector_type(16))) _Float16 v16h;
typedef __attribute__((ext_vector_type(8)))  _Float16 v8h;
typedef __attribute__((ext_vector_type(8)))  float    v8f;
typedef __attribute__((ext_vector_type(4)))  unsigned int u32x4;

#define NIMG 128
#define CIN  256
#define CH   128
#define HH   14
#define WW   14
#define HW   196
#define PWPERCONV (9*4*8*32*16)   // taps * kchunks * mtiles * lanes * halfs

// ---------------------------------------------------------------------------
// Pack conv3x3 128->128 weights [COUT][CIN][3][3] f32 into the WMMA 16-bit
// A-fragment order used in this file, contiguous k (16x32 per fragment): lanes 0-15 hold K{0..15},
// lanes 16-31 hold K{16..31} for M = lane&15.
// dst flat index = (((tap*4+kt)*8 + mtile)*32 + lane)*16 + e
// ---------------------------------------------------------------------------
__global__ void pack_w_kernel(const float* __restrict__ w0,
                              const float* __restrict__ w1,
                              const float* __restrict__ w2,
                              _Float16* __restrict__ dst) {
  const int conv = blockIdx.y;
  const float* src = (conv == 0) ? w0 : ((conv == 1) ? w1 : w2);
  _Float16* d = dst + (size_t)conv * PWPERCONV;
  int idx = blockIdx.x * blockDim.x + threadIdx.x;
  if (idx >= PWPERCONV) return;
  int e    = idx & 15;
  int lane = (idx >> 4) & 31;
  int mt   = (idx >> 9) & 7;
  int kt   = (idx >> 12) & 3;
  int tap  = idx >> 14;              // 0..8
  int koff = (lane < 16) ? e : e + 16;
  int cout = mt * 16 + (lane & 15);
  int chn  = kt * 32 + koff;
  d[idx] = (_Float16)src[(cout * CH + chn) * 9 + tap];
}

// ---------------------------------------------------------------------------
// 1x1 conv (256->9) + sigmoid + tap-normalization -> nw [N][9][HW] f32
// ---------------------------------------------------------------------------
__global__ void regnorm_kernel(const float* __restrict__ x,
                               const float* __restrict__ rw,
                               const float* __restrict__ rb,
                               float* __restrict__ nw) {
  __shared__ float sw[9 * CIN];
  __shared__ float sb[9];
  for (int i = threadIdx.x; i < 9 * CIN; i += blockDim.x) sw[i] = rw[i];
  if (threadIdx.x < 9) sb[threadIdx.x] = rb[threadIdx.x];
  __syncthreads();
  int idx = blockIdx.x * blockDim.x + threadIdx.x;
  if (idx >= NIMG * HW) return;
  int n = idx / HW, p = idx - (idx / HW) * HW;
  float acc[9];
#pragma unroll
  for (int t = 0; t < 9; ++t) acc[t] = sb[t];
  const float* xp = x + (size_t)n * CIN * HW + p;
  for (int ci = 0; ci < CIN; ++ci) {
    float xv = xp[(size_t)ci * HW];
#pragma unroll
    for (int t = 0; t < 9; ++t) acc[t] += sw[t * CIN + ci] * xv;
  }
  float s = 1e-5f;
#pragma unroll
  for (int t = 0; t < 9; ++t) {
    acc[t] = 1.0f / (1.0f + __expf(-acc[t]));
    s += acc[t];
  }
  float inv = 1.0f / s;
#pragma unroll
  for (int t = 0; t < 9; ++t)
    nw[((size_t)n * 9 + t) * HW + p] = acc[t] * inv;
}

// ---------------------------------------------------------------------------
// Encoder conv1: saliency [N][1][14][14] -> emb [N][HW][128] f16 (ReLU)
// ---------------------------------------------------------------------------
__global__ void enc1_kernel(const float* __restrict__ sal,
                            const float* __restrict__ w1,
                            const float* __restrict__ b1,
                            _Float16* __restrict__ out) {
  __shared__ float sw[CH * 9];
  __shared__ float sb[CH];
  for (int i = threadIdx.x; i < CH * 9; i += blockDim.x) sw[i] = w1[i];
  for (int i = threadIdx.x; i < CH; i += blockDim.x) sb[i] = b1[i];
  __syncthreads();
  int idx = blockIdx.x * blockDim.x + threadIdx.x;
  if (idx >= NIMG * HW) return;
  int n = idx / HW, p = idx - (idx / HW) * HW;
  int y = p / WW, xx = p - (p / WW) * WW;
  float s9[9];
#pragma unroll
  for (int t = 0; t < 9; ++t) {
    int dy = t / 3, dx = t - (t / 3) * 3;
    int ny = y + dy - 1, nx = xx + dx - 1;
    s9[t] = ((unsigned)ny < (unsigned)HH && (unsigned)nx < (unsigned)WW)
                ? sal[(size_t)n * HW + ny * WW + nx] : 0.0f;
  }
  _Float16* op = out + ((size_t)n * HW + p) * CH;
  for (int c = 0; c < CH; ++c) {
    float a = sb[c];
#pragma unroll
    for (int t = 0; t < 9; ++t) a += sw[c * 9 + t] * s9[t];
    op[c] = (_Float16)fmaxf(a, 0.0f);
  }
}

// ---------------------------------------------------------------------------
// One chunk of NT output N-tiles for the WMMA conv. Everything inlined into
// the kernel; acc pressure = NT*8 VGPRs.
// ---------------------------------------------------------------------------
template <int NT>
__device__ __forceinline__ void conv_chunk(const _Float16* __restrict__ tile,
                                           const _Float16* __restrict__ pwA,
                                           const float* __restrict__ bv,
                                           _Float16* __restrict__ outImg,
                                           int ntStart, int lanelo, int hi,
                                           int mbase, int relu) {
  int ldsoff[NT];     // element offset into tile for this lane's B column
  bool pval[NT];
#pragma unroll
  for (int j = 0; j < NT; ++j) {
    int p = (ntStart + j) * 16 + lanelo;
    pval[j] = (p < HW);
    int pc = pval[j] ? p : (HW - 1);      // clamp: garbage cols never stored
    int y = pc / WW, xx = pc - (pc / WW) * WW;
    ldsoff[j] = (y * 16 + xx) * CH + hi * 16;
  }
  v8f acc[NT];
#pragma unroll
  for (int j = 0; j < NT; ++j) acc[j] = (v8f)0.0f;

#pragma unroll 1
  for (int i = 0; i < 36; ++i) {          // i = tap*4 + kchunk
    int tap = i >> 2;
    int kt  = i & 3;
    int dy = tap / 3, dx = tap - dy * 3;
    int iterm = (dy * 16 + dx) * CH + kt * 32;   // uniform per iteration
    v16h a = *(const v16h*)(pwA + (size_t)i * (8 * 32 * 16));
#pragma unroll
    for (int j = 0; j < NT; ++j) {
      v16h b = *(const v16h*)(tile + ldsoff[j] + iterm);
      acc[j] = __builtin_amdgcn_wmma_f32_16x16x32_f16(
          false, a, false, b, (short)0, acc[j], false, false);
    }
  }

#pragma unroll
  for (int j = 0; j < NT; ++j) {
    if (!pval[j]) continue;
    int p = (ntStart + j) * 16 + lanelo;
    v8h hv;
#pragma unroll
    for (int r = 0; r < 8; ++r) {
      float v = acc[j][r] + bv[r];
      if (relu) v = fmaxf(v, 0.0f);
      hv[r] = (_Float16)v;
    }
    *(v8h*)(outImg + (size_t)p * CH + mbase) = hv;
  }
}

// ---------------------------------------------------------------------------
// conv3x3 128->128 via WMMA implicit GEMM. One block per image.
// LDS: zero-padded 16x16 spatial tile, pixel-major, 128 f16 channels / pixel.
// 8 waves = 8 cout tiles of 16. N-tiles processed in chunks of 7+6 to keep
// accumulator pressure at <=56 VGPRs; launch_bounds(256,1) frees the full
// register budget so nothing spills to scratch.
// ---------------------------------------------------------------------------
__global__ __launch_bounds__(256, 1)
void conv128_wmma_kernel(const _Float16* __restrict__ in,
                         const _Float16* __restrict__ pw,
                         const float* __restrict__ bias,
                         _Float16* __restrict__ out, int relu) {
  extern __shared__ char smem[];
  _Float16* tile = (_Float16*)smem;                       // 256*128 halfs = 64KB
  float* sb = (float*)(smem + 256 * CH * sizeof(_Float16));
  const int n = blockIdx.x;
  const int t = threadIdx.x;

  { // stage padded tile: thread t owns padded pixel t (16x16)
    int py = t >> 4, px = t & 15;
    int y = py - 1, xx = px - 1;
    u32x4* dst = (u32x4*)(tile + t * CH);
    if ((unsigned)y < (unsigned)HH && (unsigned)xx < (unsigned)WW) {
      const u32x4* src = (const u32x4*)(in + ((size_t)n * HW + y * WW + xx) * CH);
#pragma unroll
      for (int i = 0; i < 16; ++i) dst[i] = src[i];
    } else {
      u32x4 z = (u32x4)0u;
#pragma unroll
      for (int i = 0; i < 16; ++i) dst[i] = z;
    }
    if (t < CH) sb[t] = bias[t];
  }
  __syncthreads();

  const int wv = t >> 5;        // mtile (cout/16)
  const int lane = t & 31;
  const int lanelo = lane & 15;
  const int hi = lane >> 4;     // which K half this lane holds for B
  const int mbase = wv * 16 + hi * 8;

  const _Float16* pwA = pw + ((size_t)(wv * 32) + lane) * 16;
  _Float16* outImg = out + (size_t)n * HW * CH;

  float bv[8];
#pragma unroll
  for (int r = 0; r < 8; ++r) bv[r] = sb[mbase + r];

  conv_chunk<7>(tile, pwA, bv, outImg, 0, lanelo, hi, mbase, relu);
  conv_chunk<6>(tile, pwA, bv, outImg, 7, lanelo, hi, mbase, relu);
}

// ---------------------------------------------------------------------------
// Message passing: 14 iterations of spatially-varying 3x3 neighbor sum,
// entirely LDS-resident (double-buffered 64KB tiles + 7KB weights).
// In-place on emb [N][HW][128] f16. One block per image.
// ---------------------------------------------------------------------------
__global__ void prop_kernel(_Float16* __restrict__ emb,
                            const float* __restrict__ nwg) {
  extern __shared__ char smem[];
  _Float16* buf0 = (_Float16*)smem;              // 256*128
  _Float16* buf1 = buf0 + 256 * CH;              // 256*128
  float* snw = (float*)(smem + 2 * 256 * CH * sizeof(_Float16)); // 9*196
  const int n = blockIdx.x;
  const int t = threadIdx.x;

  { // zero both padded buffers (borders stay zero forever)
    u32x4 z = (u32x4)0u;
    u32x4* b = (u32x4*)buf0;
    for (int i = t; i < (2 * 256 * CH) / 8; i += 256) b[i] = z;
  }
  for (int i = t; i < 9 * HW; i += 256) snw[i] = nwg[(size_t)n * 9 * HW + i];
  __syncthreads();

  if (t < HW) { // stage interior of buf0
    int y = t / WW, xx = t - (t / WW) * WW;
    const u32x4* src = (const u32x4*)(emb + ((size_t)n * HW + t) * CH);
    u32x4* dst = (u32x4*)(buf0 + ((y + 1) * 16 + xx + 1) * CH);
#pragma unroll
    for (int i = 0; i < 16; ++i) dst[i] = src[i];
  }
  __syncthreads();

  _Float16* cur = buf0;
  _Float16* nxt = buf1;
  for (int it = 0; it < 14; ++it) {
    for (int idx = t; idx < HW * 16; idx += 256) {
      int c8 = idx & 15;          // channel chunk of 8
      int p  = idx >> 4;
      int y = p / WW, xx = p - (p / WW) * WW;
      float acc[8];
#pragma unroll
      for (int e = 0; e < 8; ++e) acc[e] = 0.0f;
#pragma unroll
      for (int tap = 0; tap < 9; ++tap) {
        int dy = tap / 3, dx = tap - (tap / 3) * 3;
        float w = snw[tap * HW + p];
        v8h v = *(const v8h*)(cur + ((y + dy) * 16 + (xx + dx)) * CH + c8 * 8);
#pragma unroll
        for (int e = 0; e < 8; ++e) acc[e] += w * (float)v[e];
      }
      v8h o;
#pragma unroll
      for (int e = 0; e < 8; ++e) o[e] = (_Float16)acc[e];
      *(v8h*)(nxt + ((y + 1) * 16 + (xx + 1)) * CH + c8 * 8) = o;
    }
    __syncthreads();
    _Float16* tmp = cur; cur = nxt; nxt = tmp;
  }

  if (t < HW) { // write interior back
    int y = t / WW, xx = t - (t / WW) * WW;
    const u32x4* src = (const u32x4*)(cur + ((y + 1) * 16 + xx + 1) * CH);
    u32x4* dst = (u32x4*)(emb + ((size_t)n * HW + t) * CH);
#pragma unroll
    for (int i = 0; i < 16; ++i) dst[i] = src[i];
  }
}

// ---------------------------------------------------------------------------
// Final conv3x3 128->1, f32 output [N][14][14]
// ---------------------------------------------------------------------------
__global__ void dec3_kernel(const _Float16* __restrict__ in,
                            const float* __restrict__ w3,
                            const float* __restrict__ b3,
                            float* __restrict__ out) {
  __shared__ float sw[CH * 9];
  for (int i = threadIdx.x; i < CH * 9; i += blockDim.x) sw[i] = w3[i];
  __syncthreads();
  int idx = blockIdx.x * blockDim.x + threadIdx.x;
  if (idx >= NIMG * HW) return;
  int n = idx / HW, p = idx - (idx / HW) * HW;
  int y = p / WW, xx = p - (p / WW) * WW;
  float acc = b3[0];
#pragma unroll 1
  for (int tap = 0; tap < 9; ++tap) {
    int dy = tap / 3, dx = tap - (tap / 3) * 3;
    int ny = y + dy - 1, nx = xx + dx - 1;
    if ((unsigned)ny >= (unsigned)HH || (unsigned)nx >= (unsigned)WW) continue;
    const v8h* src = (const v8h*)(in + ((size_t)n * HW + ny * WW + nx) * CH);
    for (int c8 = 0; c8 < 16; ++c8) {
      v8h v = src[c8];
#pragma unroll
      for (int e = 0; e < 8; ++e)
        acc += (float)v[e] * sw[(c8 * 8 + e) * 9 + tap];
    }
  }
  out[(size_t)n * HW + p] = acc;
}

// ---------------------------------------------------------------------------
extern "C" void kernel_launch(void* const* d_in, const int* in_sizes, int n_in,
                              void* d_out, int out_size, void* d_ws, size_t ws_size,
                              hipStream_t stream) {
  const float* x     = (const float*)d_in[0];
  const float* sal   = (const float*)d_in[1];
  const float* reg_w = (const float*)d_in[2];
  const float* reg_b = (const float*)d_in[3];
  const float* ew1   = (const float*)d_in[4];
  const float* eb1   = (const float*)d_in[5];
  const float* ew2   = (const float*)d_in[6];
  const float* eb2   = (const float*)d_in[7];
  const float* dw1   = (const float*)d_in[8];
  const float* db1   = (const float*)d_in[9];
  const float* dw2   = (const float*)d_in[10];
  const float* db2   = (const float*)d_in[11];
  const float* dw3   = (const float*)d_in[12];
  const float* db3   = (const float*)d_in[13];
  float* out = (float*)d_out;

  char* ws = (char*)d_ws;
  size_t off = 0;
  auto alloc = [&](size_t bytes) -> char* {
    char* p = ws + off;
    off += (bytes + 255) & ~(size_t)255;
    return p;
  };
  const size_t NW_BYTES  = (size_t)NIMG * 9 * HW * sizeof(float);
  const size_t EMB_BYTES = (size_t)NIMG * HW * CH * sizeof(_Float16);
  const size_t PW_BYTES  = (size_t)3 * PWPERCONV * sizeof(_Float16);

  float*    nw    = (float*)alloc(NW_BYTES);
  _Float16* emb_a = (_Float16*)alloc(EMB_BYTES);   // enc1 out
  _Float16* emb_b = (_Float16*)alloc(EMB_BYTES);   // enc2 out / prop in-place
  _Float16* emb_c = (_Float16*)alloc(EMB_BYTES);   // dec1 out
  _Float16* emb_d = (_Float16*)alloc(EMB_BYTES);   // dec2 out
  _Float16* pwbuf = (_Float16*)alloc(PW_BYTES);
  (void)ws_size; (void)in_sizes; (void)n_in; (void)out_size;

  const int BT = 256;
  const int PIX_BLOCKS = (NIMG * HW + BT - 1) / BT;       // 98
  const size_t CONV_SHM = (size_t)256 * CH * 2 + CH * 4;  // 66048
  const size_t PROP_SHM = (size_t)2 * 256 * CH * 2 + 9 * HW * 4; // 138128

  // 1) pack the three 128->128 conv weights into WMMA A-fragment layout
  pack_w_kernel<<<dim3(PWPERCONV / BT, 3), BT, 0, stream>>>(ew2, dw1, dw2, pwbuf);
  // 2) propagation-weight regressor + sigmoid + normalize
  regnorm_kernel<<<PIX_BLOCKS, BT, 0, stream>>>(x, reg_w, reg_b, nw);
  // 3) encoder conv1 (1->128) + ReLU
  enc1_kernel<<<PIX_BLOCKS, BT, 0, stream>>>(sal, ew1, eb1, emb_a);
  // 4) encoder conv2 (128->128), no ReLU
  conv128_wmma_kernel<<<NIMG, BT, CONV_SHM, stream>>>(
      emb_a, pwbuf + 0 * PWPERCONV, eb2, emb_b, 0);
  // 5) 14-iteration LDS-resident message passing (in-place on emb_b)
  prop_kernel<<<NIMG, BT, PROP_SHM, stream>>>(emb_b, nw);
  // 6) decoder conv1 (128->128) + ReLU
  conv128_wmma_kernel<<<NIMG, BT, CONV_SHM, stream>>>(
      emb_b, pwbuf + 1 * PWPERCONV, db1, emb_c, 1);
  // 7) decoder conv2 (128->128) + ReLU
  conv128_wmma_kernel<<<NIMG, BT, CONV_SHM, stream>>>(
      emb_c, pwbuf + 2 * PWPERCONV, db2, emb_d, 1);
  // 8) decoder conv3 (128->1), f32 out
  dec3_kernel<<<PIX_BLOCKS, BT, 0, stream>>>(emb_d, dw3, db3, out);
}
